// Sampler_63050119905678
// MI455X (gfx1250) — matrix operand order repaired and checked by reading, not compiled, not run on hardware
//
#include <hip/hip_runtime.h>
#include <hip/hip_bf16.h>

typedef __attribute__((ext_vector_type(16))) __bf16 v16bf;
typedef __attribute__((ext_vector_type(8)))  float  v8f;

#define NTOK   64
#define HID    4096
#define VOCAB  32000
#define SEPS   1e-5f
#define BK     64          // K chunk per stage
#define SA     (BK + 8)    // bf16 row stride (144B, 16B aligned)
#define SB     (BK + 8)    // f32  row stride (288B, 16B aligned)
#define NCHUNK (HID / BK)  // 64

// ---- CDNA5 async global->LDS copy (ASYNCcnt), 16B per lane ---------------
__device__ __forceinline__ void async_ld_b128(unsigned lds_byte_off,
                                              const void* gaddr) {
  asm volatile("global_load_async_to_lds_b128 %0, %1, off"
               :: "v"(lds_byte_off), "v"(gaddr) : "memory");
}
template <int N>
__device__ __forceinline__ void wait_async_le() {
#if __has_builtin(__builtin_amdgcn_s_wait_asynccnt)
  __builtin_amdgcn_s_wait_asynccnt(N);   // requires constant integer
#else
  asm volatile("s_wait_asynccnt %0" :: "n"(N) : "memory");
#endif
}
__device__ __forceinline__ unsigned lds_off32(const void* p) {
  return (unsigned)(unsigned long long)p;   // LDS aperture: low 32 bits
}

// ---------------------------------------------------------------- kernel 1
// Gather last-token rows, fold temperature (logit/T == (h/T)@E^T), cast bf16.
__global__ void gather_scale_bf16(const float* __restrict__ hs,
                                  const float* __restrict__ temps,
                                  const int*   __restrict__ lti,
                                  __bf16*      __restrict__ hbuf) {
  const int b = blockIdx.x;
  const int idx = lti[b];
  const float t = temps[b];
  const float scale = (t < SEPS) ? 1.0f : (1.0f / t);
  const float* src = hs + (size_t)idx * HID;
  __bf16* dst = hbuf + (size_t)b * HID;
  for (int d = threadIdx.x; d < HID; d += blockDim.x)
    dst[d] = (__bf16)(src[d] * scale);
}

// ---------------------------------------------------------------- kernel 2
// logits[64][32000] = hbuf(bf16) @ E^T, v_wmma_f32_16x16x32_bf16,
// async global->LDS double-buffered staging (B kept fp32 in LDS, converted
// to bf16 while building fragments).
__global__ __launch_bounds__(256)
void logits_wmma(const __bf16* __restrict__ hbuf,
                 const float*  __restrict__ emb,
                 float*        __restrict__ logits) {
  __shared__ __bf16 As[2][64][SA];   // 2*64*72*2  = 18 KB
  __shared__ float  Bsf[2][64][SB];  // 2*64*72*4  = 36 KB

  const int tid  = threadIdx.x;
  const int wid  = tid >> 5;
  const int lane = tid & 31;
  const int half = lane >> 4;
  const int lm   = lane & 15;
  const int n0   = blockIdx.x * 64;

  const int mt  = wid >> 1;        // M tile 0..3
  const int ntb = (wid & 1) * 2;   // N tile base (wave owns 2 N tiles)

  v8f acc0 = {}; v8f acc1 = {};

  // staging decomposition: thread -> (row, 16-elem column chunk)
  const int r   = tid >> 2;        // 0..63
  const int c16 = (tid & 3) * 16;  // 0,16,32,48

  // issue one chunk's async loads: A = 2x b128 (bf16), B = 4x b128 (fp32)
  auto stage = [&](int k0, int buf) {
    const __bf16* ga = hbuf + (size_t)r * HID + k0 + c16;
    async_ld_b128(lds_off32(&As[buf][r][c16]),     ga);
    async_ld_b128(lds_off32(&As[buf][r][c16 + 8]), ga + 8);
    const float* gb = emb + (size_t)(n0 + r) * HID + k0 + c16;
    async_ld_b128(lds_off32(&Bsf[buf][r][c16]),      gb);
    async_ld_b128(lds_off32(&Bsf[buf][r][c16 + 4]),  gb + 4);
    async_ld_b128(lds_off32(&Bsf[buf][r][c16 + 8]),  gb + 8);
    async_ld_b128(lds_off32(&Bsf[buf][r][c16 + 12]), gb + 12);
  };

  stage(0, 0);   // prologue prefetch

  for (int kc = 0; kc < NCHUNK; ++kc) {
    const int cur = kc & 1;
    const bool more = (kc + 1 < NCHUNK);
    if (more) {
      stage((kc + 1) * BK, cur ^ 1);
      // our 6 asyncs for chunk kc are done once count drops to the 6 just issued
      wait_async_le<6>();
    } else {
      wait_async_le<0>();
    }
    __syncthreads();

    #pragma unroll
    for (int kk = 0; kk < BK; kk += 32) {
      // A fragment: 16-bit A 16x32 layout
      v16bf a;
      {
        const int row = mt * 16 + lm;
        #pragma unroll
        for (int p = 0; p < 8; ++p) {
          const int kb = kk + (p & 3) * 2 + (p >> 2) * 16 + half * 8;
          a[2 * p]     = As[cur][row][kb];
          a[2 * p + 1] = As[cur][row][kb + 1];
        }
      }
      // B fragments: 32x16 layout; convert fp32->bf16 while loading
      v16bf b0, b1;
      {
        const int col0 = ntb * 16 + lm;
        const int kbase = kk + half * 8;
        #pragma unroll
        for (int e = 0; e < 16; ++e) {
          b0[e] = (__bf16)Bsf[cur][col0][kbase + (e & 7) + (e >> 3) * 16];
          b1[e] = (__bf16)Bsf[cur][col0 + 16][kbase + (e & 7) + (e >> 3) * 16];
        }
      }
      acc0 = __builtin_amdgcn_wmma_f32_16x16x32_bf16(false, a, false, b0,
                                                     (short)0, acc0, false, false);
      acc1 = __builtin_amdgcn_wmma_f32_16x16x32_bf16(false, a, false, b1,
                                                     (short)0, acc1, false, false);
    }
    __syncthreads();   // protect buf (cur becomes write target next iter)
  }

  // C/D layout: VGPR i -> M = i + 8*half, N = lm
  #pragma unroll
  for (int i = 0; i < 8; ++i) {
    const int row = mt * 16 + half * 8 + i;
    logits[(size_t)row * VOCAB + n0 + ntb * 16 + lm]       = acc0[i];
    logits[(size_t)row * VOCAB + n0 + (ntb + 1) * 16 + lm] = acc1[i];
  }
}

// ---------------------------------------------------------------- kernel 3
__device__ __forceinline__ float blockSum(float v, float* red) {
  red[threadIdx.x] = v; __syncthreads();
  for (int s = 256; s > 0; s >>= 1) {
    if ((int)threadIdx.x < s) red[threadIdx.x] += red[threadIdx.x + s];
    __syncthreads();
  }
  float r = red[0]; __syncthreads();
  return r;
}
__device__ __forceinline__ float blockMax(float v, float* red) {
  red[threadIdx.x] = v; __syncthreads();
  for (int s = 256; s > 0; s >>= 1) {
    if ((int)threadIdx.x < s)
      red[threadIdx.x] = fmaxf(red[threadIdx.x], red[threadIdx.x + s]);
    __syncthreads();
  }
  float r = red[0]; __syncthreads();
  return r;
}

// One block per sequence: softmax, top-k via bit-binary-search, top-p via
// conditional-sum binary search, inverse-CDF sample, write filtered probs.
__global__ __launch_bounds__(512)
void sample_row(const float* __restrict__ logits,
                const float* __restrict__ top_ps,
                const float* __restrict__ uarr,
                const int*   __restrict__ top_ks,
                float*       __restrict__ out) {
  const int b = blockIdx.x;
  const int tid = threadIdx.x;
  __shared__ float red[512];
  __shared__ float vals[512];
  __shared__ int   token_s;
  __shared__ float running_s;

  const float* lr = logits + (size_t)b * VOCAB;
  float* pr = out + NTOK + (size_t)b * VOCAB;   // probs region doubles as scratch

  // softmax
  float lmax = -3.4e38f;
  for (int v = tid; v < VOCAB; v += 512) lmax = fmaxf(lmax, lr[v]);
  lmax = blockMax(lmax, red);
  float ls = 0.f;
  for (int v = tid; v < VOCAB; v += 512) {
    float p = __expf(lr[v] - lmax);
    pr[v] = p;
    ls += p;
  }
  const float inv = 1.0f / blockSum(ls, red);
  for (int v = tid; v < VOCAB; v += 512) pr[v] *= inv;
  __syncthreads();

  // top-k threshold: kth largest prob (binary search on float bit pattern)
  int k = top_ks[b];
  k = (k < 1) ? 1 : (k > VOCAB ? VOCAB : k);
  unsigned lo = 0u, hi = 0x3F800001u;
  for (int it = 0; it < 32; ++it) {
    const unsigned mid = lo + ((hi - lo) >> 1);
    if (mid == lo) break;
    const float t = __uint_as_float(mid);
    float c = 0.f;
    for (int v = tid; v < VOCAB; v += 512) c += (pr[v] >= t) ? 1.f : 0.f;
    c = blockSum(c, red);
    if (c >= (float)k) lo = mid; else hi = mid;
  }
  const float kth = __uint_as_float(lo);

  // top-p: keep p iff sum of larger kept probs <= top_p
  const float tp = top_ps[b];
  const bool keepAll = (tp >= 1.0f - SEPS);
  float tKeep = 0.f;
  if (!keepAll) {
    unsigned lo2 = 0u, hi2 = 0x3F800001u;
    for (int it = 0; it < 32; ++it) {
      const unsigned mid = lo2 + ((hi2 - lo2) >> 1);
      if (mid == lo2) break;
      const float t = __uint_as_float(mid);
      float g = 0.f;
      for (int v = tid; v < VOCAB; v += 512) {
        const float p = pr[v];
        if (p >= kth && p > t) g += p;
      }
      g = blockSum(g, red);
      if (g <= tp) hi2 = mid; else lo2 = mid;
    }
    tKeep = __uint_as_float(hi2);
  }

  // total of filtered probs; fallback to unfiltered if all zeroed
  float s = 0.f;
  for (int v = tid; v < VOCAB; v += 512) {
    const float p = pr[v];
    if (p >= kth && (keepAll || p >= tKeep)) s += p;
  }
  float total = blockSum(s, red);
  const bool fallback = !(total > 0.f);
  if (fallback) {
    float s2 = 0.f;
    for (int v = tid; v < VOCAB; v += 512) s2 += pr[v];
    total = blockSum(s2, red);
  }

  // final pass: write f/total and inverse-CDF sample in token order
  const float target = uarr[b] * total;
  const float invT = 1.0f / total;
  if (tid == 0) { token_s = -1; running_s = 0.f; }
  __syncthreads();
  for (int base = 0; base < VOCAB; base += 512) {
    const int v = base + tid;
    const float p = (v < VOCAB) ? pr[v] : 0.f;
    const float fv = fallback ? p
                   : ((p >= kth && (keepAll || p >= tKeep)) ? p : 0.f);
    vals[tid] = fv;
    const float csum = blockSum(fv, red);   // barrier also publishes vals
    if (tid == 0 && token_s < 0) {
      if (running_s + csum >= target && csum > 0.f) {
        float acc = running_s;
        for (int i = 0; i < 512; ++i) {
          acc += vals[i];
          if (acc >= target) { token_s = base + i; break; }
        }
        if (token_s < 0) token_s = base + 511;
      } else {
        running_s += csum;
      }
    }
    __syncthreads();
    if (v < VOCAB) pr[v] = fv * invT;
  }
  if (tid == 0) {
    int tok = token_s;
    if (tok < 0 || tok > VOCAB - 1) tok = VOCAB - 1;
    out[b] = (float)tok;
  }
}

// ---------------------------------------------------------------- launcher
extern "C" void kernel_launch(void* const* d_in, const int* in_sizes, int n_in,
                              void* d_out, int out_size, void* d_ws, size_t ws_size,
                              hipStream_t stream) {
  const float* hs    = (const float*)d_in[0];
  const float* emb   = (const float*)d_in[1];
  const float* temps = (const float*)d_in[2];
  const float* tps   = (const float*)d_in[3];
  const float* u     = (const float*)d_in[4];
  const int*   lti   = (const int*)d_in[5];
  const int*   tks   = (const int*)d_in[6];
  float* out = (float*)d_out;

  __bf16* hbuf  = (__bf16*)d_ws;                                   // 512 KB
  float*  logit = (float*)((char*)d_ws + (size_t)NTOK * HID * 2);  // 8 MB

  gather_scale_bf16<<<NTOK, 256, 0, stream>>>(hs, temps, lti, hbuf);
  logits_wmma<<<VOCAB / 64, 256, 0, stream>>>(hbuf, emb, logit);
  sample_row<<<NTOK, 512, 0, stream>>>(logit, tps, u, tks, out);
}
